// MultiHeadSelfAttention_1382979470007
// MI455X (gfx1250) — matrix operand order repaired and checked by reading, not compiled, not run on hardware
//
#include <hip/hip_runtime.h>
#include <stdint.h>

// ---------------------------------------------------------------------------
// MI455X (gfx1250) multi-head self-attention, wave32 + WMMA f16 pipeline.
// B=2 T=2048 D=1024 H=16 HD=64.  ~69 GFLOP dense, working set < L2 (192MB)
// => matrix-core bound => all GEMM/attention math via v_wmma_f32_16x16x32_f16,
//    tiles staged with CDNA5 async global->LDS (ASYNCcnt) instead of VGPR bounce.
// ---------------------------------------------------------------------------

typedef __attribute__((ext_vector_type(16))) _Float16 v16h;
typedef __attribute__((ext_vector_type(8)))  _Float16 v8h;
typedef __attribute__((ext_vector_type(4)))  _Float16 v4h;
typedef __attribute__((ext_vector_type(8)))  float    v8f;
typedef __attribute__((ext_vector_type(4)))  float    v4f;

#define B_SZ 2
#define T_SZ 2048
#define D_SZ 1024
#define H_SZ 16
#define HD_SZ 64
#define M_SZ (B_SZ * T_SZ)   // 4096 rows

__device__ __forceinline__ v16h mk16(v8h lo, v8h hi) {
  return __builtin_shufflevector(lo, hi, 0, 1, 2, 3, 4, 5, 6, 7,
                                 8, 9, 10, 11, 12, 13, 14, 15);
}

// A-matrix 16x32 f16 fragment as loaded here: per lane row = lane&15,
// halfwords [0:15] = K = kg*16..kg*16+15 (contiguous, matches B fragments).
__device__ __forceinline__ v16h load_afrag(const _Float16* rowptr, int kg) {
  v8h lo = *(const v8h*)(rowptr + kg * 16);
  v8h hi = *(const v8h*)(rowptr + kg * 16 + 8);
  return mk16(lo, hi);
}

__device__ __forceinline__ v8f wmma16(v16h a, v16h b, v8f c) {
  return __builtin_amdgcn_wmma_f32_16x16x32_f16(
      /*neg_a=*/false, a, /*neg_b=*/false, b,
      /*c_mod=*/(short)0, c, /*reuse_a=*/false, /*reuse_b=*/false);
}

// CDNA5 async global->LDS copy of 16 bytes.  VDST carries the LDS byte
// offset: generic LDS pointers hold the offset in their low 32 bits
// (aperture is in [63:32], flat-addressing rules).
__device__ __forceinline__ void async_g2l_b128(_Float16* lds,
                                               const _Float16* gptr) {
  unsigned lds_off = (unsigned)(uintptr_t)lds;
  unsigned long long ga = (unsigned long long)(uintptr_t)gptr;
  asm volatile("global_load_async_to_lds_b128 %0, %1, off"
               :: "v"(lds_off), "v"(ga) : "memory");
}

__device__ __forceinline__ void wait_async0() {
  asm volatile("s_wait_asynccnt 0x0" ::: "memory");
}

// ---------------------------------------------------------------------------
// fp32 -> fp16 convert, 4 elems/thread
// ---------------------------------------------------------------------------
__global__ void cvt_f32_f16(const v4f* __restrict__ in, v4h* __restrict__ out,
                            int n4) {
  int i = blockIdx.x * blockDim.x + threadIdx.x;
  if (i < n4) {
    v4f a = in[i];
    v4h h;
#pragma unroll
    for (int j = 0; j < 4; ++j) h[j] = (_Float16)a[j];
    out[i] = h;
  }
}

// ---------------------------------------------------------------------------
// C[M,N] = A[M,K] * B[N,K]^T  (both f16 row-major, K contiguous == x @ W.T)
// block: 128 threads (4 wave32), 64(M) x 128(N) C tile, K stepped by 32.
// Per wave: 1 A-fragment reused across 8 B-fragments -> 8 WMMAs / k-step.
// Tiles staged with global_load_async_to_lds_b128 (ASYNCcnt).
// f32out=0: f16 output.  f32out=1: f32 output + bias.
// ---------------------------------------------------------------------------
__global__ __launch_bounds__(128) void gemm_nt_wmma(
    const _Float16* __restrict__ A, const _Float16* __restrict__ Bm,
    _Float16* __restrict__ Ch, float* __restrict__ Cf,
    const float* __restrict__ bias, int M, int N, int K, int f32out) {
  __shared__ __align__(32) _Float16 As[64 * 32];
  __shared__ __align__(32) _Float16 Bs[128 * 32];

  const int tid = threadIdx.x;
  const int lane = tid & 31, wave = tid >> 5;
  const int r = lane & 15, kg = lane >> 4;
  const int bm = blockIdx.y * 64, bn = blockIdx.x * 128;

  v8f acc[8] = {};

  for (int k0 = 0; k0 < K; k0 += 32) {
    // A tile: 64 rows x 4 chunks = 256 ; B tile: 128 rows x 4 chunks = 512
#pragma unroll
    for (int i = 0; i < 2; ++i) {
      int idx = tid + i * 128;           // 0..255
      int row = idx >> 2, seg = idx & 3; // seg: 8-halfword segment
      async_g2l_b128(As + row * 32 + seg * 8,
                     A + (size_t)(bm + row) * K + k0 + seg * 8);
    }
#pragma unroll
    for (int i = 0; i < 4; ++i) {
      int idx = tid + i * 128;           // 0..511
      int row = idx >> 2, seg = idx & 3;
      async_g2l_b128(Bs + row * 32 + seg * 8,
                     Bm + (size_t)(bn + row) * K + k0 + seg * 8);
    }
    wait_async0();
    __syncthreads();

    if (k0 + 32 < K) {  // -> global_prefetch_b8 for the next K tile
      int row = tid >> 1;
      __builtin_prefetch(A + (size_t)(bm + (row & 63)) * K + k0 + 32, 0, 1);
      __builtin_prefetch(Bm + (size_t)(bn + row) * K + k0 + 32, 0, 1);
    }

    v16h af = load_afrag(As + (wave * 16 + r) * 32, kg);
#pragma unroll
    for (int g = 0; g < 8; ++g) {
      // B 32x16 fragment: lane = col n, 16 consecutive K at kg*16
      v16h bf = *(const v16h*)(Bs + (g * 16 + r) * 32 + kg * 16);
      acc[g] = wmma16(af, bf, acc[g]);
    }
    __syncthreads();
  }

  // C layout: VGPR e -> rows e (lanes 0-15) / e+8 (lanes 16-31), lane = N
#pragma unroll
  for (int g = 0; g < 8; ++g) {
#pragma unroll
    for (int e = 0; e < 8; ++e) {
      int row = bm + wave * 16 + e + 8 * kg;
      int col = bn + g * 16 + r;
      if (f32out)
        Cf[(size_t)row * N + col] = acc[g][e] + bias[col];
      else
        Ch[(size_t)row * N + col] = (_Float16)acc[g][e];
    }
  }
}

// ---------------------------------------------------------------------------
// Flash attention (causal).  grid = (B*H, T/64), block = 128 (4 waves).
// Each wave owns 16 query rows; per key tile of 64:
//   scores:  8 WMMAs (4 N-groups x 2 K-chunks over HD=64)
//   softmax: wave32 xor-butterfly inside 16-lane halves + running m/l
//   P @ V:   8 WMMAs, P restaged via per-wave LDS (C-layout -> A-layout)
// K tile staged via async global->LDS; V transposed on the way in (VGPR path).
// ---------------------------------------------------------------------------
__global__ __launch_bounds__(128) void attn_wmma(
    const _Float16* __restrict__ Q, const _Float16* __restrict__ Km,
    const _Float16* __restrict__ V, _Float16* __restrict__ O) {
  __shared__ __align__(32) _Float16 Ks[64 * 64];       // [key][d]
  __shared__ __align__(32) _Float16 Vt[64 * 64];       // [d][key] (transposed)
  __shared__ __align__(32) _Float16 Ps[4][16 * 64];    // per-wave P staging

  const int tid = threadIdx.x;
  const int lane = tid & 31, wave = tid >> 5;
  const int r = lane & 15, kg = lane >> 4;

  const int bh = blockIdx.x;
  const int b = bh / H_SZ, h = bh % H_SZ;
  const int qb = blockIdx.y;

  const size_t rowbase = (size_t)b * T_SZ;
  const int colbase = h * HD_SZ;
  const int qrow = qb * 64 + wave * 16 + r;

  // Q fragments (two 16x32 chunks over HD=64), straight from global
  v16h qf[2];
#pragma unroll
  for (int c = 0; c < 2; ++c)
    qf[c] = load_afrag(Q + (rowbase + qrow) * D_SZ + colbase + c * 32, kg);

  float m_run[8], l_run[8];
#pragma unroll
  for (int e = 0; e < 8; ++e) { m_run[e] = -3.0e38f; l_run[e] = 0.0f; }
  v8f o_acc[4] = {};

  const float scale = 0.125f;  // HD^-0.5

  for (int j = 0; j <= qb; ++j) {
    // ---- stage K (async) and V (transposed, VGPR path) tiles into LDS ----
#pragma unroll
    for (int i = 0; i < 4; ++i) {
      int idx = tid + i * 128;            // 0..511
      int krow = idx >> 3, seg = idx & 7; // 8 segments of 8 halfwords
      size_t grow = (rowbase + j * 64 + krow) * D_SZ + colbase + seg * 8;
      async_g2l_b128(Ks + krow * 64 + seg * 8, Km + grow);
      v8h vv = *(const v8h*)(V + grow);
#pragma unroll
      for (int e = 0; e < 8; ++e) Vt[(seg * 8 + e) * 64 + krow] = vv[e];
    }
    wait_async0();
    __syncthreads();

    // ---- scores S = Q K^T (16 q-rows x 64 keys) ----
    v8f s[4] = {};
#pragma unroll
    for (int g = 0; g < 4; ++g) {
#pragma unroll
      for (int c = 0; c < 2; ++c) {
        v16h kf = *(const v16h*)(Ks + (g * 16 + r) * 64 + c * 32 + kg * 16);
        s[g] = wmma16(qf[c], kf, s[g]);
      }
    }

    // ---- scale + causal mask ----
#pragma unroll
    for (int g = 0; g < 4; ++g) {
#pragma unroll
      for (int e = 0; e < 8; ++e) {
        float val = s[g][e] * scale;
        int qi = qb * 64 + wave * 16 + e + 8 * kg;
        int kj = j * 64 + g * 16 + r;
        s[g][e] = (kj > qi) ? -3.0e38f : val;
      }
    }

    // ---- online softmax (per row e; xor masks 1..8 stay in 16-lane half) ----
#pragma unroll
    for (int e = 0; e < 8; ++e) {
      float mx = fmaxf(fmaxf(s[0][e], s[1][e]), fmaxf(s[2][e], s[3][e]));
#pragma unroll
      for (int m = 1; m < 16; m <<= 1) mx = fmaxf(mx, __shfl_xor(mx, m, 32));
      float mnew = fmaxf(m_run[e], mx);
      float corr = __expf(m_run[e] - mnew);
      m_run[e] = mnew;
      float rs = 0.0f;
#pragma unroll
      for (int g = 0; g < 4; ++g) {
        float p = __expf(s[g][e] - mnew);
        s[g][e] = p;
        rs += p;
      }
#pragma unroll
      for (int m = 1; m < 16; m <<= 1) rs += __shfl_xor(rs, m, 32);
      l_run[e] = l_run[e] * corr + rs;
#pragma unroll
      for (int g = 0; g < 4; ++g) o_acc[g][e] *= corr;
    }

    // ---- restage P as f16 A-operand via per-wave LDS ----
    _Float16* ps = &Ps[wave][0];
#pragma unroll
    for (int g = 0; g < 4; ++g) {
#pragma unroll
      for (int e = 0; e < 8; ++e)
        ps[(e + 8 * kg) * 64 + g * 16 + r] = (_Float16)s[g][e];
    }

    // ---- O += P @ V  (A = P 16x64, B = Vt chunks 32x16) ----
#pragma unroll
    for (int c = 0; c < 2; ++c) {
      v16h pf = load_afrag(ps + r * 64 + c * 32, kg);
#pragma unroll
      for (int g = 0; g < 4; ++g) {
        v16h vf = *(const v16h*)(Vt + (g * 16 + r) * 64 + c * 32 + kg * 16);
        o_acc[g] = wmma16(pf, vf, o_acc[g]);
      }
    }
    __syncthreads();  // before overwriting Ks/Vt next tile
  }

  // ---- normalize and store f16 head output ----
#pragma unroll
  for (int e = 0; e < 8; ++e) {
    float inv = 1.0f / l_run[e];
#pragma unroll
    for (int g = 0; g < 4; ++g) {
      int row = qb * 64 + wave * 16 + e + 8 * kg;
      int col = colbase + g * 16 + r;
      O[(rowbase + row) * D_SZ + col] = (_Float16)(o_acc[g][e] * inv);
    }
  }
}

// ---------------------------------------------------------------------------
// launcher
// ---------------------------------------------------------------------------
extern "C" void kernel_launch(void* const* d_in, const int* in_sizes, int n_in,
                              void* d_out, int out_size, void* d_ws,
                              size_t ws_size, hipStream_t stream) {
  const float* x  = (const float*)d_in[0];
  const float* Wk = (const float*)d_in[1];
  const float* Wq = (const float*)d_in[2];
  const float* Wv = (const float*)d_in[3];
  const float* Wp = (const float*)d_in[4];
  const float* bp = (const float*)d_in[5];
  float* out = (float*)d_out;

  // workspace layout (halfwords), all 256B-aligned by construction
  char* ws = (char*)d_ws;
  _Float16* xh  = (_Float16*)(ws);                              //  8 MB
  _Float16* Wkh = (_Float16*)(ws + (size_t)8  * 1024 * 1024);   //  2 MB
  _Float16* Wqh = (_Float16*)(ws + (size_t)10 * 1024 * 1024);   //  2 MB
  _Float16* Wvh = (_Float16*)(ws + (size_t)12 * 1024 * 1024);   //  2 MB
  _Float16* Wph = (_Float16*)(ws + (size_t)14 * 1024 * 1024);   //  2 MB
  _Float16* Qh  = (_Float16*)(ws + (size_t)16 * 1024 * 1024);   //  8 MB
  _Float16* Kh  = (_Float16*)(ws + (size_t)24 * 1024 * 1024);   //  8 MB
  _Float16* Vh  = (_Float16*)(ws + (size_t)32 * 1024 * 1024);   //  8 MB
  _Float16* Ah  = (_Float16*)(ws + (size_t)40 * 1024 * 1024);   //  8 MB

  const int M = M_SZ, N = D_SZ, K = D_SZ;

  // 1) fp32 -> fp16
  {
    int n4 = (M * D_SZ) / 4;
    cvt_f32_f16<<<n4 / 256, 256, 0, stream>>>((const v4f*)x, (v4h*)xh, n4);
    int w4 = (D_SZ * D_SZ) / 4;
    cvt_f32_f16<<<w4 / 256, 256, 0, stream>>>((const v4f*)Wk, (v4h*)Wkh, w4);
    cvt_f32_f16<<<w4 / 256, 256, 0, stream>>>((const v4f*)Wq, (v4h*)Wqh, w4);
    cvt_f32_f16<<<w4 / 256, 256, 0, stream>>>((const v4f*)Wv, (v4h*)Wvh, w4);
    cvt_f32_f16<<<w4 / 256, 256, 0, stream>>>((const v4f*)Wp, (v4h*)Wph, w4);
  }

  // 2) Q/K/V projections: Y = xh @ W^T   (f16 out)
  dim3 ggrid(N / 128, M / 64);
  gemm_nt_wmma<<<ggrid, 128, 0, stream>>>(xh, Wqh, Qh, nullptr, nullptr,
                                          M, N, K, 0);
  gemm_nt_wmma<<<ggrid, 128, 0, stream>>>(xh, Wkh, Kh, nullptr, nullptr,
                                          M, N, K, 0);
  gemm_nt_wmma<<<ggrid, 128, 0, stream>>>(xh, Wvh, Vh, nullptr, nullptr,
                                          M, N, K, 0);

  // 3) causal flash attention per (b,h), 64-query tiles
  attn_wmma<<<dim3(B_SZ * H_SZ, T_SZ / 64), 128, 0, stream>>>(Qh, Kh, Vh, Ah);

  // 4) output projection + bias (f32 out)
  gemm_nt_wmma<<<ggrid, 128, 0, stream>>>(Ah, Wph, nullptr, out, bp,
                                          M, N, K, 1);
}
